// TinyAttention_14285061226676
// MI455X (gfx1250) — matrix operand order repaired and checked by reading, not compiled, not run on hardware
//
#include <hip/hip_runtime.h>
#include <hip/hip_bf16.h>
#include <cstdint>

typedef __attribute__((ext_vector_type(16))) _Float16 v16h;
typedef __attribute__((ext_vector_type(8)))  float    v8f;

union frag16 { v16h h; uint4 q[2]; };

#define BATCH   2
#define SEQ     2048
#define DIM     768
#define HEADS   12
#define HDIM    64
#define MTOT    (BATCH*SEQ)          // 4096
#define SCALE_Q 0.125f               // 64^-0.5
#define L2E     1.44269504f

// lane <-> lane^16 exchange, pure VALU (v_permlanex16_b32)
__device__ __forceinline__ float xor16f(float x)
{
    int i = __builtin_bit_cast(int, x);
    int r = __builtin_amdgcn_permlanex16(i, i, 0x76543210, 0xfedcba98,
                                         false, false);
    return __builtin_bit_cast(float, r);
}

// byte offset of an LDS location (generic -> AS(3) -> u32)
__device__ __forceinline__ unsigned lds_addr(const void* p)
{
    return (unsigned)(unsigned long long)
           (const __attribute__((address_space(3))) void*)p;
}

// async DMA: 16 bytes global -> LDS per lane, tracked by ASYNCcnt
__device__ __forceinline__ void async_b128(unsigned lds_off, const void* gaddr)
{
    asm volatile("global_load_async_to_lds_b128 %0, %1, off"
                 :: "v"(lds_off), "v"(gaddr) : "memory");
}

__device__ __forceinline__ void wait_async0()
{
    asm volatile("s_wait_asynccnt 0" ::: "memory");
}

// ---------------------------------------------------------------------------
// 1) f32 -> f16 convert for x
// ---------------------------------------------------------------------------
__global__ __launch_bounds__(256)
void cvt_x_f16(const float4* __restrict__ src, uint2* __restrict__ dst)
{
    int i = blockIdx.x * 256 + threadIdx.x;
    float4 v = src[i];
    union { _Float16 h[4]; uint2 u; } o;
    o.h[0] = (_Float16)v.x; o.h[1] = (_Float16)v.y;
    o.h[2] = (_Float16)v.z; o.h[3] = (_Float16)v.w;
    dst[i] = o.u;
}

// ---------------------------------------------------------------------------
// 2) weight transpose + convert: Wt[n*DIM + k] = (f16) W[k*DIM + n]
// ---------------------------------------------------------------------------
__global__ __launch_bounds__(256)
void cvt_w_transpose(const float* __restrict__ W, _Float16* __restrict__ Wt)
{
    __shared__ float tile[32][33];
    int tx = threadIdx.x, ty = threadIdx.y;       // 32 x 8
    int bx = blockIdx.x, by = blockIdx.y;
    #pragma unroll
    for (int i = 0; i < 4; ++i)
        tile[ty + 8*i][tx] = W[(by*32 + ty + 8*i)*DIM + bx*32 + tx];
    __syncthreads();
    #pragma unroll
    for (int i = 0; i < 4; ++i)
        Wt[(bx*32 + ty + 8*i)*DIM + by*32 + tx] = (_Float16)tile[tx][ty + 8*i];
}

// ---------------------------------------------------------------------------
// 3) WMMA GEMM: C[4096 x 768] = A * Bt^T + bias. Tile 128x128, 8 waves of
//    32x64, BK=32. Async global->LDS double-buffered pipeline (ASYNCcnt).
//    MODE 0: -> qh [B,H,S,Dh] f16, (c+bias)*SCALE_Q
//    MODE 1: -> kh [B,H,S,Dh] f16
//    MODE 2: -> vt [B,H,Dh,S] f16 transposed
//    MODE 3: -> out [M,768] f32 direct
// ---------------------------------------------------------------------------
template<int MODE>
__global__ __launch_bounds__(256)
void gemm_wmma(const _Float16* __restrict__ A,
               const _Float16* __restrict__ Bt,
               const float*    __restrict__ bias,
               void*           __restrict__ Out)
{
    // two buffers of (A:128x48 + B:128x48) f16; epilogue Ct (128x136) reuses
    __shared__ alignas(16) char smem[2*2*128*48*2];     // 49152 B
    _Float16* Ct = (_Float16*)smem;

    const int bm   = blockIdx.x;                  // 0..31
    const int bn   = blockIdx.y;                  // 0..5
    const int t    = threadIdx.x;
    const int wave = t >> 5, lane = t & 31;
    const int l16  = lane & 15, lh = lane >> 4;
    const int wm   = (wave & 3) * 32;
    const int wn   = (wave >> 2) * 64;

    v8f zf = {};
    v8f acc[2][4];
    #pragma unroll
    for (int i = 0; i < 2; ++i)
        #pragma unroll
        for (int j = 0; j < 4; ++j) acc[i][j] = zf;

    int rowA[2], segA[2];
    #pragma unroll
    for (int r = 0; r < 2; ++r) { int i = t + 256*r; rowA[r] = i >> 2; segA[r] = i & 3; }

    auto Abuf = [&](int b) { return (_Float16*)(smem + b*24576); };
    auto Bbuf = [&](int b) { return (_Float16*)(smem + b*24576 + 12288); };

    auto issue = [&](int buf, int k0) {
        #pragma unroll
        for (int r = 0; r < 2; ++r) {
            async_b128(lds_addr(&Abuf(buf)[rowA[r]*48 + segA[r]*8]),
                       &A [(size_t)(bm*128 + rowA[r])*DIM + k0 + segA[r]*8]);
            async_b128(lds_addr(&Bbuf(buf)[rowA[r]*48 + segA[r]*8]),
                       &Bt[(size_t)(bn*128 + rowA[r])*DIM + k0 + segA[r]*8]);
        }
    };

    issue(0, 0);
    for (int it = 0; it < DIM/32; ++it) {
        const int k0 = it*32, cur = it & 1;
        wait_async0();
        __syncthreads();                           // tile(cur) fully in LDS
        if (k0 + 32 < DIM) issue(cur ^ 1, k0 + 32);

        const _Float16* Asm = Abuf(cur);
        const _Float16* Bsm = Bbuf(cur);
        frag16 af[2];
        #pragma unroll
        for (int mi = 0; mi < 2; ++mi) {
            const _Float16* p = &Asm[(wm + 16*mi + l16)*48 + 8*lh];
            af[mi].q[0] = *(const uint4*)p;
            af[mi].q[1] = *(const uint4*)(p + 16);
        }
        #pragma unroll
        for (int ni = 0; ni < 4; ++ni) {
            frag16 bf;
            const _Float16* p = &Bsm[(wn + 16*ni + l16)*48 + 8*lh];
            bf.q[0] = *(const uint4*)p;
            bf.q[1] = *(const uint4*)(p + 16);
            #pragma unroll
            for (int mi = 0; mi < 2; ++mi)
                acc[mi][ni] = __builtin_amdgcn_wmma_f32_16x16x32_f16(
                    false, af[mi].h, false, bf.h, (short)0, acc[mi][ni],
                    false, false);
        }
    }

    if (MODE == 3) {
        float* O = (float*)Out;
        #pragma unroll
        for (int mi = 0; mi < 2; ++mi)
            #pragma unroll
            for (int ni = 0; ni < 4; ++ni) {
                int n = bn*128 + wn + 16*ni + l16;
                float bv = bias[n];
                #pragma unroll
                for (int r = 0; r < 8; ++r) {
                    int m = bm*128 + wm + 16*mi + r + 8*lh;
                    O[(size_t)m*DIM + n] = acc[mi][ni][r] + bv;
                }
            }
        return;
    }

    __syncthreads();                               // all tile reads done
    const float scale = (MODE == 0) ? SCALE_Q : 1.0f;
    #pragma unroll
    for (int mi = 0; mi < 2; ++mi)
        #pragma unroll
        for (int ni = 0; ni < 4; ++ni) {
            int nl = wn + 16*ni + l16;
            float bv = bias[bn*128 + nl];
            #pragma unroll
            for (int r = 0; r < 8; ++r) {
                int ml = wm + 16*mi + r + 8*lh;
                _Float16 hv = (_Float16)((acc[mi][ni][r] + bv) * scale);
                if (MODE == 2) Ct[nl*136 + ml] = hv;   // [n][m]
                else           Ct[ml*136 + nl] = hv;   // [m][n]
            }
        }
    __syncthreads();

    _Float16* O = (_Float16*)Out;
    const int b = (bm*128) >> 11;
    #pragma unroll
    for (int j = 0; j < 8; ++j) {
        int i   = t + 256*j;
        int row = i >> 4, seg = i & 15;
        uint4 val = *(uint4*)&Ct[row*136 + seg*8];
        if (MODE == 2) {
            int n = bn*128 + row, h = n >> 6, d = n & 63;
            int s = (bm*128 + seg*8) & 2047;
            *(uint4*)&O[((size_t)(b*HEADS + h)*HDIM + d)*SEQ + s] = val;
        } else {
            int m = bm*128 + row, s = m & 2047;
            int n = bn*128 + seg*8, h = n >> 6, d = n & 63;
            *(uint4*)&O[((size_t)(b*HEADS + h)*SEQ + s)*HDIM + d] = val;
        }
    }
}

// ---------------------------------------------------------------------------
// 4) Flash attention, transposed orientation, async double-buffered K/V:
//      St = K_tile @ Q^T    (query rows live in lanes)
//      O^T = V^T @ P^T      (softmax stats are per-lane scalars)
// ---------------------------------------------------------------------------
__global__ __launch_bounds__(128)
void attn_flash(const _Float16* __restrict__ qh,
                const _Float16* __restrict__ kh,
                const _Float16* __restrict__ vt,
                _Float16*       __restrict__ oh)
{
    __shared__ alignas(16) _Float16 Ks[2][64*72];     // [key][d]
    __shared__ alignas(16) _Float16 Vs[2][64*72];     // [d][key]

    const int bh   = blockIdx.y;                  // b*HEADS + h
    const int qb   = blockIdx.x;                  // 0..31
    const int t    = threadIdx.x;
    const int wave = t >> 5, lane = t & 31;
    const int l16  = lane & 15, lh = lane >> 4;

    // Q^T B-fragments straight from global (n=qrow in lane, k=d interleaved)
    const _Float16* qp = qh + ((size_t)bh*SEQ + qb*64 + wave*16)*HDIM;
    frag16 qf[2];
    #pragma unroll
    for (int kk = 0; kk < 2; ++kk) {
        const _Float16* p = qp + l16*HDIM + 32*kk + 8*lh;
        qf[kk].q[0] = *(const uint4*)p;
        qf[kk].q[1] = *(const uint4*)(p + 16);
    }

    v8f zf = {};
    v8f acc[4];
    #pragma unroll
    for (int i = 0; i < 4; ++i) acc[i] = zf;
    float m_i = -3.0e38f, l_i = 0.f;

    int rowT[4], segT[4];
    #pragma unroll
    for (int r = 0; r < 4; ++r) { int i = t + 128*r; rowT[r] = i >> 3; segT[r] = i & 7; }

    const _Float16* kbase = kh + (size_t)bh*SEQ*HDIM;
    const _Float16* vbase = vt + (size_t)bh*HDIM*SEQ;

    auto issue = [&](int buf, int key0) {
        #pragma unroll
        for (int r = 0; r < 4; ++r) {
            async_b128(lds_addr(&Ks[buf][rowT[r]*72 + segT[r]*8]),
                       &kbase[(size_t)(key0 + rowT[r])*HDIM + segT[r]*8]);
            async_b128(lds_addr(&Vs[buf][rowT[r]*72 + segT[r]*8]),
                       &vbase[(size_t)rowT[r]*SEQ + key0 + segT[r]*8]);
        }
    };

    issue(0, 0);
    for (int it = 0; it < SEQ/64; ++it) {
        const int key0 = it*64, cur = it & 1;
        wait_async0();
        __syncthreads();                           // K/V tile(cur) ready
        if (key0 + 64 < SEQ) issue(cur ^ 1, key0 + 64);

        // St = K @ Q^T : 4 key-subtiles x 2 d-chunks
        v8f sc[4];
        #pragma unroll
        for (int mi = 0; mi < 4; ++mi) {
            sc[mi] = zf;
            #pragma unroll
            for (int kk = 0; kk < 2; ++kk) {
                frag16 af;                        // A: row=key, k=d interleaved
                const _Float16* p = &Ks[cur][(16*mi + l16)*72 + 32*kk + 8*lh];
                af.q[0] = *(const uint4*)p;
                af.q[1] = *(const uint4*)(p + 16);
                sc[mi] = __builtin_amdgcn_wmma_f32_16x16x32_f16(
                    false, af.h, false, qf[kk].h, (short)0, sc[mi],
                    false, false);
            }
        }

        // online softmax: per-lane stats + one lane-half combine
        float pm = -3.0e38f;
        #pragma unroll
        for (int mi = 0; mi < 4; ++mi)
            #pragma unroll
            for (int r = 0; r < 8; ++r) pm = fmaxf(pm, sc[mi][r]);
        pm = fmaxf(pm, xor16f(pm));
        float mn = fmaxf(m_i, pm);
        float alpha = __builtin_amdgcn_exp2f((m_i - mn) * L2E);
        m_i = mn;
        float rs = 0.f;
        #pragma unroll
        for (int mi = 0; mi < 4; ++mi)
            #pragma unroll
            for (int r = 0; r < 8; ++r) {
                float p = __builtin_amdgcn_exp2f((sc[mi][r] - mn) * L2E);
                sc[mi][r] = p;
                rs += p;
            }
        rs += xor16f(rs);
        l_i = l_i * alpha + rs;
        #pragma unroll
        for (int mi = 0; mi < 4; ++mi) acc[mi] *= alpha;

        // P^T B-fragments via permlanex16 exchange
        frag16 pf[2];
        #pragma unroll
        for (int kk = 0; kk < 2; ++kk) {
            #pragma unroll
            for (int r = 0; r < 8; ++r) {
                float send = lh ? sc[2*kk][r] : sc[2*kk+1][r];
                float recv = xor16f(send);
                float lo = lh ? recv : sc[2*kk][r];
                float hi = lh ? sc[2*kk+1][r] : recv;
                pf[kk].h[r]     = (_Float16)lo;
                pf[kk].h[8 + r] = (_Float16)hi;
            }
        }

        // O^T += V^T @ P^T : 4 d-subtiles x 2 key-chunks
        #pragma unroll
        for (int mi = 0; mi < 4; ++mi) {
            #pragma unroll
            for (int kk = 0; kk < 2; ++kk) {
                frag16 af;                        // A: row=d, k=key contiguous
                const _Float16* p = &Vs[cur][(16*mi + l16)*72 + 32*kk + 16*lh];
                af.q[0] = *(const uint4*)p;
                af.q[1] = *(const uint4*)(p + 8);
                acc[mi] = __builtin_amdgcn_wmma_f32_16x16x32_f16(
                    false, af.h, false, pf[kk].h, (short)0, acc[mi],
                    false, false);
            }
        }
    }

    // normalize + coalesced b128 stores: lane = qrow, vgprs = contiguous d
    const int b = bh / HEADS, h = bh % HEADS;
    const int s = qb*64 + wave*16 + l16;
    const float linv = 1.0f / l_i;
    _Float16* obase = oh + ((size_t)(b*SEQ + s))*DIM + h*HDIM + 8*lh;
    #pragma unroll
    for (int mi = 0; mi < 4; ++mi) {
        union { _Float16 h[8]; uint4 q; } ov;
        #pragma unroll
        for (int r = 0; r < 8; ++r) ov.h[r] = (_Float16)(acc[mi][r] * linv);
        *(uint4*)&obase[16*mi] = ov.q;
    }
}

// ---------------------------------------------------------------------------
// launcher
// ---------------------------------------------------------------------------
extern "C" void kernel_launch(void* const* d_in, const int* in_sizes, int n_in,
                              void* d_out, int out_size, void* d_ws, size_t ws_size,
                              hipStream_t stream)
{
    const float* x  = (const float*)d_in[0];
    const float* Wq = (const float*)d_in[1];
    const float* bq = (const float*)d_in[2];
    const float* Wk = (const float*)d_in[3];
    const float* bk = (const float*)d_in[4];
    const float* Wv = (const float*)d_in[5];
    const float* bv = (const float*)d_in[6];
    const float* Wo = (const float*)d_in[7];
    const float* bo = (const float*)d_in[8];
    float* out = (float*)d_out;

    char* ws = (char*)d_ws;
    const size_t SZ_X = (size_t)MTOT*DIM*2;
    const size_t SZ_W = (size_t)DIM*DIM*2;
    _Float16* xh  = (_Float16*)(ws);
    _Float16* wqt = (_Float16*)(ws + SZ_X);
    _Float16* wkt = (_Float16*)(ws + SZ_X + SZ_W);
    _Float16* wvt = (_Float16*)(ws + SZ_X + 2*SZ_W);
    _Float16* wot = (_Float16*)(ws + SZ_X + 3*SZ_W);
    _Float16* qh  = (_Float16*)(ws + SZ_X + 4*SZ_W);
    _Float16* kh  = (_Float16*)(ws + 2*SZ_X + 4*SZ_W);
    _Float16* vt  = (_Float16*)(ws + 3*SZ_X + 4*SZ_W);
    _Float16* oh  = (_Float16*)(ws + 4*SZ_X + 4*SZ_W);

    cvt_x_f16<<<(MTOT*DIM/4)/256, 256, 0, stream>>>(
        (const float4*)x, (uint2*)xh);

    dim3 tb(32, 8), tg(DIM/32, DIM/32);
    cvt_w_transpose<<<tg, tb, 0, stream>>>(Wq, wqt);
    cvt_w_transpose<<<tg, tb, 0, stream>>>(Wk, wkt);
    cvt_w_transpose<<<tg, tb, 0, stream>>>(Wv, wvt);
    cvt_w_transpose<<<tg, tb, 0, stream>>>(Wo, wot);

    dim3 gg(MTOT/128, DIM/128);
    gemm_wmma<0><<<gg, 256, 0, stream>>>(xh, wqt, bq, (void*)qh);
    gemm_wmma<1><<<gg, 256, 0, stream>>>(xh, wkt, bk, (void*)kh);
    gemm_wmma<2><<<gg, 256, 0, stream>>>(xh, wvt, bv, (void*)vt);

    dim3 ag(SEQ/64, BATCH*HEADS);
    attn_flash<<<ag, 128, 0, stream>>>(qh, kh, vt, oh);

    gemm_wmma<3><<<gg, 256, 0, stream>>>(oh, wot, bo, (void*)out);
}
